// ISODformer_20435454395198
// MI455X (gfx1250) — matrix operand order repaired and checked by reading, not compiled, not run on hardware
//
#include <hip/hip_runtime.h>
#include <hip/hip_bf16.h>
#include <math.h>

// ---------------------------------------------------------------------------
// ISODformer forward, algebraically collapsed (no softmax in reference =>
// out = (scale * V_s K^T) Q, and Wout folds in: Y = [Wout blockdiag(S)] Q).
// B=4, C=512, H=W=56 (M=3136), stride-2 grid 28x28 (N=784, padded to 800),
// 8 heads x 64 ch.
// ---------------------------------------------------------------------------

#define BATCH 4
#define CCH   512
#define HH    56
#define WW    56
#define HWPX  3136
#define HK    28
#define WKD   28
#define NKV   784
#define NKVP  800        // padded to a multiple of the WMMA K-step (32)
#define NH    8
#define HC    64

typedef __attribute__((ext_vector_type(16))) __bf16 v16bf;
typedef __attribute__((ext_vector_type(8)))  __bf16 v8bf;
typedef __attribute__((ext_vector_type(8)))  float  v8f;
typedef __attribute__((ext_vector_type(4)))  int    v4i;

#if defined(__AMDGCN__) && __has_builtin(__builtin_amdgcn_global_load_async_to_lds_b128) && __has_builtin(__builtin_amdgcn_s_wait_asynccnt)
#define HAS_ASYNC_LDS 1
// Builtin signature (from hipcc diagnostic): param0 = v4i in AS1 ("__device__"),
// param1 = v4i in AS3, then imm offset + imm cpol.
#define GAS(p) ((__attribute__((address_space(1))) v4i*)(p))
#define LAS(p) ((__attribute__((address_space(3))) v4i*)(p))
#else
#define HAS_ASYNC_LDS 0
#endif

// ---------------------------------------------------------------------------
// Kernel 1: fused depthwise 3x3 convs. One read of the x neighborhood feeds
// all four filters. q is written bf16 *transposed* ([b][m][c], the GEMM
// B-operand layout), k as padded bf16, v/off_pre fp32.
// ---------------------------------------------------------------------------
__global__ __launch_bounds__(256) void fused_dwconv(
    const float* __restrict__ x,
    const float* __restrict__ Wv, const float* __restrict__ bv,
    const float* __restrict__ Wq, const float* __restrict__ bq,
    const float* __restrict__ Wk, const float* __restrict__ bk,
    const float* __restrict__ Wo, const float* __restrict__ bo,
    __bf16* __restrict__ qbf, float* __restrict__ vout,
    __bf16* __restrict__ kout, float* __restrict__ oout)
{
  int idx = blockIdx.x * 256 + threadIdx.x;
  if (idx >= BATCH * CCH * HWPX) return;
  int w  = idx % WW;
  int h  = (idx / WW) % HH;
  int hw = idx % HWPX;
  int c  = (idx / HWPX) % CCH;
  int b  = idx / (CCH * HWPX);
  const float* xp = x + (size_t)(idx / HWPX) * HWPX;   // [b][c] plane

  float n[9];
#pragma unroll
  for (int dy = -1; dy <= 1; ++dy) {
#pragma unroll
    for (int dx = -1; dx <= 1; ++dx) {
      int hh = h + dy, ww = w + dx;
      n[(dy + 1) * 3 + (dx + 1)] =
          (hh >= 0 && hh < HH && ww >= 0 && ww < WW) ? xp[hh * WW + ww] : 0.f;
    }
  }

  const float* wv = Wv + c * 9;
  const float* wq = Wq + c * 9;
  const float* wk = Wk + c * 9;
  const float* wo = Wo + c * 9;
  float sv = bv[c], sq = bq[c], sk = bk[c], so = bo[c];
#pragma unroll
  for (int t = 0; t < 9; ++t) {
    sv += n[t] * wv[t];
    sq += n[t] * wq[t];
    sk += n[t] * wk[t];
    so += n[t] * wo[t];
  }

  vout[idx] = sv;
  qbf[((size_t)b * HWPX + hw) * CCH + c] = (__bf16)sq;   // [b][m][c]
  if (((h & 1) == 0) && ((w & 1) == 0)) {                // stride-2 taps
    int p = (h >> 1) * WKD + (w >> 1);
    size_t o = (size_t)(b * CCH + c) * NKVP + p;         // padded pitch
    kout[o] = (__bf16)sk;
    oout[(size_t)(b * CCH + c) * NKV + p] = so;
  }
}

// Zero the K-dim padding of k and v_s so head_outer needs no bounds checks.
__global__ __launch_bounds__(256) void zero_tail(
    __bf16* __restrict__ km, __bf16* __restrict__ vsb)
{
  int i = blockIdx.x * 256 + threadIdx.x;
  if (i >= BATCH * CCH * (NKVP - NKV)) return;
  int lane = i % (NKVP - NKV);
  int row  = i / (NKVP - NKV);
  size_t o = (size_t)row * NKVP + NKV + lane;
  km[o]  = (__bf16)0.f;
  vsb[o] = (__bf16)0.f;
}

// ---------------------------------------------------------------------------
// Kernel 2: per (b, p) channel-LayerNorm -> GELU(exact) -> 1x1 to 2ch offset,
// tanh(off + ref) -> bilinear sample of value => v_s[b][c][p] (bf16, padded).
// ---------------------------------------------------------------------------
__device__ __forceinline__ float gelu_exact(float v) {
  return 0.5f * v * (1.0f + erff(v * 0.70710678118654752f));
}

__global__ __launch_bounds__(256) void offset_sample(
    const float* __restrict__ opre, const float* __restrict__ ln_g,
    const float* __restrict__ ln_b, const float* __restrict__ Woff2,
    const float* __restrict__ value, __bf16* __restrict__ vs)
{
  int p = blockIdx.x;            // 0..783
  int b = blockIdx.y;
  int hk = p / WKD, wk = p % WKD;
  int t = threadIdx.x;

  __shared__ float rs[256];
  __shared__ float rq[256];
  __shared__ float geom[8];

  int c0 = t, c1 = t + 256;
  float x0 = opre[((size_t)(b * CCH + c0)) * NKV + p];
  float x1 = opre[((size_t)(b * CCH + c1)) * NKV + p];

  rs[t] = x0 + x1;
  rq[t] = x0 * x0 + x1 * x1;
  __syncthreads();
  for (int off = 128; off > 0; off >>= 1) {
    if (t < off) { rs[t] += rs[t + off]; rq[t] += rq[t + off]; }
    __syncthreads();
  }
  if (t == 0) {
    float mu  = rs[0] * (1.0f / CCH);
    float var = rq[0] * (1.0f / CCH) - mu * mu;
    geom[6] = mu;
    geom[7] = rsqrtf(var + 1e-5f);
  }
  __syncthreads();
  float mu = geom[6], rstd = geom[7];

  float g0 = gelu_exact((x0 - mu) * rstd * ln_g[c0] + ln_b[c0]);
  float g1 = gelu_exact((x1 - mu) * rstd * ln_g[c1] + ln_b[c1]);
  rs[t] = g0 * Woff2[c0] + g1 * Woff2[c1];              // off_y
  rq[t] = g0 * Woff2[CCH + c0] + g1 * Woff2[CCH + c1];  // off_x
  __syncthreads();
  for (int off = 128; off > 0; off >>= 1) {
    if (t < off) { rs[t] += rs[t + off]; rq[t] += rq[t + off]; }
    __syncthreads();
  }
  if (t == 0) {
    float ry = ((hk + 0.5f) / HK) * 2.f - 1.f;
    float rx = ((wk + 0.5f) / WKD) * 2.f - 1.f;
    float py = tanhf(rs[0] + ry);
    float px = tanhf(rq[0] + rx);
    float ix = (px + 1.f) * 0.5f * (WW - 1);
    float iy = (py + 1.f) * 0.5f * (HH - 1);
    float ix0f = floorf(ix), iy0f = floorf(iy);
    geom[0] = ix - ix0f;   // wx
    geom[1] = iy - iy0f;   // wy
    ((int*)geom)[2] = min(max((int)ix0f, 0), WW - 1);
    ((int*)geom)[3] = min(max((int)iy0f, 0), HH - 1);
    ((int*)geom)[4] = min(max((int)ix0f + 1, 0), WW - 1);
    ((int*)geom)[5] = min(max((int)iy0f + 1, 0), HH - 1);
  }
  __syncthreads();
  float wx = geom[0], wy = geom[1];
  int ix0 = ((int*)geom)[2], iy0 = ((int*)geom)[3];
  int ix1 = ((int*)geom)[4], iy1 = ((int*)geom)[5];

#pragma unroll
  for (int r = 0; r < 2; ++r) {
    int c = t + r * 256;
    const float* vp = value + ((size_t)(b * CCH + c)) * HWPX;
    float v00 = vp[iy0 * WW + ix0];
    float v01 = vp[iy0 * WW + ix1];
    float v10 = vp[iy1 * WW + ix0];
    float v11 = vp[iy1 * WW + ix1];
    float res = v00 * (1 - wx) * (1 - wy) + v01 * wx * (1 - wy) +
                v10 * (1 - wx) * wy + v11 * wx * wy;
    vs[((size_t)(b * CCH + c)) * NKVP + p] = (__bf16)res;
  }
}

// ---------------------------------------------------------------------------
// Kernel 3: S[b][h] = scale * V_s K^T  (64x64, K-dim = 800 padded) via WMMA
// bf16. Operands are bf16 and row-contiguous along K => branch-free 16B
// vector loads straight from global, fragments packed per the ISA layouts.
// ---------------------------------------------------------------------------
__global__ __launch_bounds__(256) void head_outer(
    const __bf16* __restrict__ vs, const __bf16* __restrict__ kmat,
    float* __restrict__ S)
{
  int bh = blockIdx.x;                 // b*8 + h
  int wid = threadIdx.x >> 5, lane = threadIdx.x & 31;
  int lm = lane & 15, khalf = lane >> 4;

  int t0 = wid * 2;                    // two adjacent column tiles per wave
  int r  = t0 >> 2;                    // A row tile 0..3
  int ca = (t0 & 3), cb = ca + 1;      // B col tiles

  const __bf16* Vrow = vs   + ((size_t)bh * HC + r * 16 + lm) * NKVP;
  const __bf16* KrA  = kmat + ((size_t)bh * HC + ca * 16 + lm) * NKVP;
  const __bf16* KrB  = kmat + ((size_t)bh * HC + cb * 16 + lm) * NKVP;

  v8f acc0 = {}, acc1 = {};

  for (int n0 = 0; n0 < NKVP; n0 += 32) {
    // A frag: lane holds K = khalf*8..+8 and 16+khalf*8..+8
    v8bf a0 = *(const v8bf*)(Vrow + n0 + khalf * 8);
    v8bf a1 = *(const v8bf*)(Vrow + n0 + 16 + khalf * 8);
    // B frags: lane holds K = khalf*8..+8 and 16+khalf*8..+8
    v8bf p0 = *(const v8bf*)(KrA + n0 + khalf * 8);
    v8bf p1 = *(const v8bf*)(KrA + n0 + 16 + khalf * 8);
    v8bf q0 = *(const v8bf*)(KrB + n0 + khalf * 8);
    v8bf q1 = *(const v8bf*)(KrB + n0 + 16 + khalf * 8);
    v16bf a, b0, b1;
#pragma unroll
    for (int j = 0; j < 8; ++j) {
      a[j] = a0[j];  a[8 + j] = a1[j];
      b0[j] = p0[j]; b0[8 + j] = p1[j];
      b1[j] = q0[j]; b1[8 + j] = q1[j];
    }
    acc0 = __builtin_amdgcn_wmma_f32_16x16x32_bf16(false, a, false, b0,
                                                   (short)0, acc0, false, false);
    acc1 = __builtin_amdgcn_wmma_f32_16x16x32_bf16(false, a, false, b1,
                                                   (short)0, acc1, false, false);
  }

  const float scale = 0.125f;          // hc^-0.5 = 1/8
  float* Sp = S + (size_t)bh * HC * HC;
#pragma unroll
  for (int v = 0; v < 8; ++v) {        // C/D layout: M = v + khalf*8, N = lm
    int row = r * 16 + v + khalf * 8;
    Sp[row * HC + ca * 16 + lm] = acc0[v] * scale;
    Sp[row * HC + cb * 16 + lm] = acc1[v] * scale;
  }
}

// ---------------------------------------------------------------------------
// Kernel 4: T[b] = Wout * blockdiag(S[b][h])  (512x512, bf16 out). Tiny.
// ---------------------------------------------------------------------------
__global__ __launch_bounds__(256) void compose_T(
    const float* __restrict__ Wout, const float* __restrict__ S,
    __bf16* __restrict__ Tbf)
{
  int idx = blockIdx.x * 256 + threadIdx.x;   // b*C*C + o*C + c
  if (idx >= BATCH * CCH * CCH) return;
  int c = idx % CCH;
  int o = (idx / CCH) % CCH;
  int b = idx / (CCH * CCH);
  int h = c >> 6, cl = c & 63;
  const float* Sp = S + ((size_t)(b * NH + h)) * HC * HC;
  const float* wp = Wout + (size_t)o * CCH + h * HC;
  float acc = 0.f;
#pragma unroll 8
  for (int j = 0; j < HC; ++j) acc += wp[j] * Sp[j * HC + cl];
  Tbf[idx] = (__bf16)acc;
}

// ---------------------------------------------------------------------------
// Kernel 5: Y[b] = T[b] (512x512) x Qt[b] (3136x512, [m][c]) -> fp32 accum
// via WMMA bf16. 128x64 block tile, 8 waves each owning 32x32 (2x2 WMMA).
// Double-buffered ASYNCcnt pipeline: tile T+1's global->LDS async copies are
// in flight while tile T is multiplied; s_wait_asynccnt(3) retires only the
// older tile (async ops complete in order), overlapping copy with math.
// ---------------------------------------------------------------------------
__global__ __launch_bounds__(256) void gemm_y(
    const __bf16* __restrict__ Tbf, const __bf16* __restrict__ Qt,
    float* __restrict__ y)
{
  int nb = blockIdx.x;   // 0..48  (3136 = 49*64)
  int mb = blockIdx.y;   // 0..3   (512 = 4*128)
  int b  = blockIdx.z;

  const __bf16* Tp = Tbf + (size_t)b * CCH * CCH + (size_t)mb * 128 * CCH;
  const __bf16* Qp = Qt + (size_t)b * HWPX * CCH + (size_t)nb * 64 * CCH;

  __shared__ __align__(16) __bf16 As[2][128][40];   // [buf][o][k], padded pitch
  __shared__ __align__(16) __bf16 Bs[2][64][40];    // [buf][n][k], padded pitch

  int t = threadIdx.x, wid = t >> 5, lane = t & 31;
  int wm = (wid & 3) * 32, wn = (wid >> 2) * 32;
  int lm = lane & 15, khalf = lane >> 4;

  // Per-thread tile-fill coordinates (all 16B-aligned lanes).
  int arow = t >> 1,  aco = (t & 1) * 16;        // A: 128 rows x 32 k
  int brow = t >> 2,  bco = (t & 3) * 8;         // B:  64 rows x 32 k

  v8f acc[2][2] = {};

  auto issue_tile = [&](int k0, int pb) {
    const __bf16* ga = Tp + (size_t)arow * CCH + k0 + aco;
    const __bf16* gb = Qp + (size_t)brow * CCH + k0 + bco;
#if HAS_ASYNC_LDS
    __builtin_amdgcn_global_load_async_to_lds_b128(GAS(ga),     LAS(&As[pb][arow][aco]),     0, 0);
    __builtin_amdgcn_global_load_async_to_lds_b128(GAS(ga + 8), LAS(&As[pb][arow][aco + 8]), 0, 0);
    __builtin_amdgcn_global_load_async_to_lds_b128(GAS(gb),     LAS(&Bs[pb][brow][bco]),     0, 0);
#else
    *(v8bf*)&As[pb][arow][aco]     = *(const v8bf*)ga;
    *(v8bf*)&As[pb][arow][aco + 8] = *(const v8bf*)(ga + 8);
    *(v8bf*)&Bs[pb][brow][bco]     = *(const v8bf*)gb;
#endif
  };

  auto compute_tile = [&](int pb) {
#pragma unroll
    for (int i = 0; i < 2; ++i) {
      v16bf a;
      int row = wm + i * 16 + lm;
      v8bf a0 = *(const v8bf*)&As[pb][row][khalf * 8];
      v8bf a1 = *(const v8bf*)&As[pb][row][16 + khalf * 8];
#pragma unroll
      for (int j = 0; j < 8; ++j) { a[j] = a0[j]; a[8 + j] = a1[j]; }
#pragma unroll
      for (int jj = 0; jj < 2; ++jj) {
        v16bf bb;
        int col = wn + jj * 16 + lm;
        v8bf b0 = *(const v8bf*)&Bs[pb][col][khalf * 8];
        v8bf b1 = *(const v8bf*)&Bs[pb][col][16 + khalf * 8];
#pragma unroll
        for (int j = 0; j < 8; ++j) { bb[j] = b0[j]; bb[8 + j] = b1[j]; }
        acc[i][jj] = __builtin_amdgcn_wmma_f32_16x16x32_bf16(
            false, a, false, bb, (short)0, acc[i][jj], false, false);
      }
    }
  };

  issue_tile(0, 0);
  int buf = 0;
  for (int k0 = 0; k0 < CCH - 32; k0 += 32) {
    issue_tile(k0 + 32, buf ^ 1);        // next tile into the other buffer
#if HAS_ASYNC_LDS
    __builtin_amdgcn_s_wait_asynccnt(3); // retire tile k0's 3 copies only
#endif
    __syncthreads();
    compute_tile(buf);
    __syncthreads();                     // readers done before buf is refilled
    buf ^= 1;
  }
#if HAS_ASYNC_LDS
  __builtin_amdgcn_s_wait_asynccnt(0);   // drain the final tile
#endif
  __syncthreads();
  compute_tile(buf);

  float* yp = y + (size_t)b * CCH * HWPX + (size_t)(mb * 128) * HWPX + nb * 64;
#pragma unroll
  for (int i = 0; i < 2; ++i)
#pragma unroll
    for (int jj = 0; jj < 2; ++jj)
#pragma unroll
      for (int v = 0; v < 8; ++v) {
        int row = wm + i * 16 + v + khalf * 8;
        int col = wn + jj * 16 + lm;
        yp[(size_t)row * HWPX + col] = acc[i][jj][v];
      }
}

// ---------------------------------------------------------------------------
extern "C" void kernel_launch(void* const* d_in, const int* in_sizes, int n_in,
                              void* d_out, int out_size, void* d_ws, size_t ws_size,
                              hipStream_t stream) {
  const float* x     = (const float*)d_in[0];
  const float* Wv    = (const float*)d_in[1];
  const float* bv    = (const float*)d_in[2];
  const float* Wq    = (const float*)d_in[3];
  const float* bq    = (const float*)d_in[4];
  const float* Wk    = (const float*)d_in[5];
  const float* bk    = (const float*)d_in[6];
  const float* Woff1 = (const float*)d_in[7];
  const float* boff1 = (const float*)d_in[8];
  const float* ln_g  = (const float*)d_in[9];
  const float* ln_b  = (const float*)d_in[10];
  const float* Woff2 = (const float*)d_in[11];
  const float* Wout  = (const float*)d_in[12];
  float* y = (float*)d_out;

  char* w = (char*)d_ws;
  __bf16* qbf = (__bf16*)w;  w += (size_t)BATCH * HWPX * CCH * 2;   // [b][m][c]
  float*  v   = (float*)w;   w += (size_t)BATCH * CCH * HWPX * 4;
  __bf16* km  = (__bf16*)w;  w += (size_t)BATCH * CCH * NKVP * 2;   // padded
  float*  op  = (float*)w;   w += (size_t)BATCH * CCH * NKV * 4;
  __bf16* vsb = (__bf16*)w;  w += (size_t)BATCH * CCH * NKVP * 2;   // padded
  float*  S   = (float*)w;   w += (size_t)BATCH * NH * HC * HC * 4;
  __bf16* Tbf = (__bf16*)w;

  int total = BATCH * CCH * HWPX;
  fused_dwconv<<<(total + 255) / 256, 256, 0, stream>>>(
      x, Wv, bv, Wq, bq, Wk, bk, Woff1, boff1, qbf, v, km, op);

  int tail = BATCH * CCH * (NKVP - NKV);
  zero_tail<<<(tail + 255) / 256, 256, 0, stream>>>(km, vsb);

  offset_sample<<<dim3(NKV, BATCH), 256, 0, stream>>>(
      op, ln_g, ln_b, Woff2, v, vsb);

  head_outer<<<BATCH * NH, 256, 0, stream>>>(vsb, km, S);

  compose_T<<<(BATCH * CCH * CCH + 255) / 256, 256, 0, stream>>>(Wout, S, Tbf);

  gemm_y<<<dim3(49, 4, BATCH), 256, 0, stream>>>(Tbf, qbf, y);
}
